// MoELayer_74371653697641
// MI455X (gfx1250) — matrix operand order repaired and checked by reading, not compiled, not run on hardware
//
#include <hip/hip_runtime.h>
#include <hip/hip_bf16.h>
#include <cstdint>

// MoE layer for MI455X (gfx1250, wave32, WMMA).
// fp32 inputs cast once to bf16; all GEMMs run on v_wmma_f32_16x16x32_bf16 with
// fp32 accumulation. Weights (~54 MB bf16) are L2-resident (192 MB L2), so WMMA
// fragments are loaded directly from global as contiguous B128 chunks matching
// a k order shared by A and B (k = 16*hi + s, not the ISA table order). Per-wave tiles are 64x64 (16 WMMAs per K-step,
// ~33 FLOP/byte from L2) to keep the compute-bound loop fed.

typedef __bf16 bf16;
typedef __attribute__((ext_vector_type(16))) __bf16 v16bf;
typedef __attribute__((ext_vector_type(8)))  __bf16 v8bf;
typedef __attribute__((ext_vector_type(8)))  float  v8f;

// Problem constants (fixed by the reference).
#define TT 4096   // tokens = B*L
#define DD 1024
#define HH 512
#define EE 16
#define SS 2048

// ---------------- fragment loads (contiguous k = 16*hi + s, same for A and B) ----------------
// A 16x32 bf16: lane m=lane&15 is row M; hi=lane>>4 selects K-half.
// elements 0..15 <- A[m, k0 + hi*16 .. +15]
__device__ __forceinline__ v16bf load_a_frag(const bf16* __restrict__ A, int lda,
                                             int row, int k0, int m16, int hi) {
  const bf16* p = A + (size_t)(row + m16) * lda + (k0 + hi * 16);
  v16bf a;
  ((v8bf*)&a)[0] = *(const v8bf*)(p);
  ((v8bf*)&a)[1] = *(const v8bf*)(p + 8);
  return a;
}

// B 32x16 bf16 (B stored row-major [N,K]): lane n=lane&15 is column N;
// elements 0..15 <- B[n, k0 + hi*16 .. +15]  (one contiguous 32-byte load)
__device__ __forceinline__ v16bf load_b_frag(const bf16* __restrict__ B, int ldb,
                                             int col, int k0, int m16, int hi) {
  return *(const v16bf*)(B + (size_t)(col + m16) * ldb + (k0 + hi * 16));
}

// silu with single-instruction v_rcp_f32 (no IEEE div_scale expansion);
// plenty of precision ahead of a bf16 store.
__device__ __forceinline__ float silu_f(float v) {
  return v * __builtin_amdgcn_rcpf(1.f + __expf(-v));
}

// ---------------- fp32 -> bf16 cast ----------------
__global__ void cast_f32_bf16(const float* __restrict__ in, bf16* __restrict__ out, int n) {
  int i = blockIdx.x * blockDim.x + threadIdx.x;
  int stride = gridDim.x * blockDim.x;
  for (; i < n; i += stride) out[i] = (bf16)in[i];
}

// ---------------- router: logits, top-2, softmax -> dense dispatch [T,E] ----------------
__global__ void router_kernel(const float* __restrict__ x, const float* __restrict__ gw,
                              float* __restrict__ dispatch) {
  __shared__ float logits[4][EE];
  int lt  = threadIdx.x;          // 64 threads = 4 tokens x 16 experts
  int sub = lt >> 4;
  int e   = lt & 15;
  int tok = blockIdx.x * 4 + sub;
  const float4* xr = (const float4*)(x + (size_t)tok * DD);
  const float4* gr = (const float4*)(gw + (size_t)e * DD);
  float acc = 0.f;
  for (int k = 0; k < DD / 4; ++k) {
    float4 a = xr[k], b = gr[k];
    acc += a.x * b.x + a.y * b.y + a.z * b.z + a.w * b.w;
  }
  logits[sub][e] = acc;
  __syncthreads();
  if (e == 0) {
    float* L = logits[sub];
    int i0 = 0; float v0 = L[0];
    for (int i = 1; i < EE; ++i) if (L[i] > v0) { v0 = L[i]; i0 = i; }
    int i1 = -1; float v1 = -INFINITY;
    for (int i = 0; i < EE; ++i) if (i != i0 && L[i] > v1) { v1 = L[i]; i1 = i; }
    float e0 = 1.f;                    // exp(v0 - v0)
    float e1 = __expf(v1 - v0);
    float inv = __builtin_amdgcn_rcpf(e0 + e1);
    float* drow = dispatch + (size_t)tok * EE;
    for (int i = 0; i < EE; ++i) drow[i] = 0.f;
    drow[i0] = e0 * inv;
    drow[i1] = e1 * inv;
  }
}

// ---------------- generic WMMA GEMM: C[m,n] = sum_k A[m,k]*B[n,k] ----------------
// EPI 0: store fp32     EPI 1: accumulate fp32 (+=)
// EPI 2: bf16 store of silu(acc) * rowscale[row*sstride]   (fused SwiGLU + dispatch)
template<int MT, int NT, int EPI>
__global__ void __launch_bounds__(256)
gemm_bf16(const bf16* __restrict__ A, const bf16* __restrict__ B, void* __restrict__ outv,
          const float* __restrict__ rowscale, int sstride,
          int M, int N, int K, int lda, int ldb, int ldc) {
  int wave = (blockIdx.x * blockDim.x + threadIdx.x) >> 5;
  int lane = threadIdx.x & 31;
  int m16 = lane & 15, hi = lane >> 4;
  int tilesN = N / (NT * 16);
  int tm = wave / tilesN, tn = wave % tilesN;
  int row0 = tm * MT * 16, col0 = tn * NT * 16;
  if (row0 >= M) return;

  v8f zero = {};
  v8f acc[MT][NT];
#pragma unroll
  for (int mi = 0; mi < MT; ++mi)
#pragma unroll
    for (int ni = 0; ni < NT; ++ni) acc[mi][ni] = zero;

  for (int k0 = 0; k0 < K; k0 += 32) {
    v16bf afr[MT], bfr[NT];
#pragma unroll
    for (int mi = 0; mi < MT; ++mi) afr[mi] = load_a_frag(A, lda, row0 + mi * 16, k0, m16, hi);
#pragma unroll
    for (int ni = 0; ni < NT; ++ni) bfr[ni] = load_b_frag(B, ldb, col0 + ni * 16, k0, m16, hi);
    // hint next K-slice of B into cache (global_prefetch_b8)
    __builtin_prefetch((const void*)(B + (size_t)(col0 + m16) * ldb + k0 + 32), 0, 1);
#pragma unroll
    for (int mi = 0; mi < MT; ++mi)
#pragma unroll
      for (int ni = 0; ni < NT; ++ni)
        acc[mi][ni] = __builtin_amdgcn_wmma_f32_16x16x32_bf16(
            false, afr[mi], false, bfr[ni], (short)0, acc[mi][ni], false, false);
  }

  // C/D layout: VGPR r -> row = r + hi*8, col = lane&15
#pragma unroll
  for (int mi = 0; mi < MT; ++mi) {
#pragma unroll
    for (int ni = 0; ni < NT; ++ni) {
#pragma unroll
      for (int r = 0; r < 8; ++r) {
        int row = row0 + mi * 16 + r + hi * 8;
        int col = col0 + ni * 16 + m16;
        float v = acc[mi][ni][r];
        if (EPI == 0) {
          ((float*)outv)[(size_t)row * ldc + col] = v;
        } else if (EPI == 1) {
          float* o = (float*)outv + (size_t)row * ldc + col;
          *o = *o + v;
        } else {
          float g = silu_f(v) * rowscale[(size_t)row * sstride];
          ((bf16*)outv)[(size_t)row * ldc + col] = (bf16)g;
        }
      }
    }
  }
}

// ---------------- gated GEMM: out_bf16 = silu(A@B1^T) * (A@B2^T) ----------------
template<int MT, int NT>
__global__ void __launch_bounds__(256)
gemm_gated_bf16(const bf16* __restrict__ A, const bf16* __restrict__ B1,
                const bf16* __restrict__ B2, bf16* __restrict__ out,
                int M, int N, int K, int lda, int ldb, int ldc) {
  int wave = (blockIdx.x * blockDim.x + threadIdx.x) >> 5;
  int lane = threadIdx.x & 31;
  int m16 = lane & 15, hi = lane >> 4;
  int tilesN = N / (NT * 16);
  int tm = wave / tilesN, tn = wave % tilesN;
  int row0 = tm * MT * 16, col0 = tn * NT * 16;
  if (row0 >= M) return;

  v8f zero = {};
  v8f acc1[MT][NT], acc2[MT][NT];
#pragma unroll
  for (int mi = 0; mi < MT; ++mi)
#pragma unroll
    for (int ni = 0; ni < NT; ++ni) { acc1[mi][ni] = zero; acc2[mi][ni] = zero; }

  for (int k0 = 0; k0 < K; k0 += 32) {
    v16bf afr[MT], b1fr[NT], b2fr[NT];
#pragma unroll
    for (int mi = 0; mi < MT; ++mi) afr[mi] = load_a_frag(A, lda, row0 + mi * 16, k0, m16, hi);
#pragma unroll
    for (int ni = 0; ni < NT; ++ni) {
      b1fr[ni] = load_b_frag(B1, ldb, col0 + ni * 16, k0, m16, hi);
      b2fr[ni] = load_b_frag(B2, ldb, col0 + ni * 16, k0, m16, hi);
    }
#pragma unroll
    for (int mi = 0; mi < MT; ++mi)
#pragma unroll
      for (int ni = 0; ni < NT; ++ni) {
        acc1[mi][ni] = __builtin_amdgcn_wmma_f32_16x16x32_bf16(
            false, afr[mi], false, b1fr[ni], (short)0, acc1[mi][ni], false, false);
        acc2[mi][ni] = __builtin_amdgcn_wmma_f32_16x16x32_bf16(
            false, afr[mi], false, b2fr[ni], (short)0, acc2[mi][ni], false, false);
      }
  }

#pragma unroll
  for (int mi = 0; mi < MT; ++mi) {
#pragma unroll
    for (int ni = 0; ni < NT; ++ni) {
#pragma unroll
      for (int r = 0; r < 8; ++r) {
        int row = row0 + mi * 16 + r + hi * 8;
        int col = col0 + ni * 16 + m16;
        float g = silu_f(acc1[mi][ni][r]) * acc2[mi][ni][r];
        out[(size_t)row * ldc + col] = (bf16)g;
      }
    }
  }
}

// ---------------- host orchestration ----------------
extern "C" void kernel_launch(void* const* d_in, const int* in_sizes, int n_in,
                              void* d_out, int out_size, void* d_ws, size_t ws_size,
                              hipStream_t stream) {
  (void)in_sizes; (void)n_in; (void)out_size; (void)ws_size;
  const float* x    = (const float*)d_in[0];  // [T, D]
  const float* gw   = (const float*)d_in[1];  // [E, D]
  const float* w1   = (const float*)d_in[2];  // [E, H, D]
  const float* w2   = (const float*)d_in[3];  // [E, D, H]
  const float* sfc1 = (const float*)d_in[4];  // [S, D]
  const float* sfc2 = (const float*)d_in[5];  // [S, D]
  const float* sfc3 = (const float*)d_in[6];  // [D, S]
  float* out = (float*)d_out;                 // [T, D]

  char* ws = (char*)d_ws;
  size_t off = 0;
  auto alloc = [&](size_t bytes) -> void* {
    void* p = ws + off;
    off += (bytes + 255) & ~size_t(255);
    return p;
  };
  bf16* x_bf    = (bf16*)alloc((size_t)TT * DD * 2);
  bf16* sfc1_bf = (bf16*)alloc((size_t)SS * DD * 2);
  bf16* sfc2_bf = (bf16*)alloc((size_t)SS * DD * 2);
  bf16* sfc3_bf = (bf16*)alloc((size_t)DD * SS * 2);
  bf16* w1_bf   = (bf16*)alloc((size_t)EE * HH * DD * 2);
  bf16* w2_bf   = (bf16*)alloc((size_t)EE * DD * HH * 2);
  float* dispatch = (float*)alloc((size_t)TT * EE * 4);
  bf16* act_bf  = (bf16*)alloc((size_t)TT * SS * 2);   // silu(x@sfc1^T)*(x@sfc2^T)
  bf16* h_bf    = (bf16*)alloc((size_t)TT * HH * 2);   // per-expert hidden, reused

  // 1) casts to bf16
  cast_f32_bf16<<<1024, 256, 0, stream>>>(x,    x_bf,    TT * DD);
  cast_f32_bf16<<<1024, 256, 0, stream>>>(sfc1, sfc1_bf, SS * DD);
  cast_f32_bf16<<<1024, 256, 0, stream>>>(sfc2, sfc2_bf, SS * DD);
  cast_f32_bf16<<<1024, 256, 0, stream>>>(sfc3, sfc3_bf, DD * SS);
  cast_f32_bf16<<<2048, 256, 0, stream>>>(w1,   w1_bf,   EE * HH * DD);
  cast_f32_bf16<<<2048, 256, 0, stream>>>(w2,   w2_bf,   EE * DD * HH);

  // 2) router -> dense dispatch weights [T, E]
  router_kernel<<<TT / 4, 64, 0, stream>>>(x, gw, dispatch);

  // 3) shared expert up (gated): act = silu(x@sfc1^T) * (x@sfc2^T)   [T, S] bf16
  {
    int waves = (TT / 32) * (SS / 64);   // MT=2, NT=4 -> 32x64 tiles (dual accum)
    gemm_gated_bf16<2, 4><<<waves / 8, 256, 0, stream>>>(
        x_bf, sfc1_bf, sfc2_bf, act_bf, TT, SS, DD, DD, DD, SS);
  }

  // 4) shared expert down: out = act @ sfc3^T   (initializes d_out, fp32)
  {
    int waves = (TT / 64) * (DD / 64);   // MT=4, NT=4 -> 64x64 tiles
    gemm_bf16<4, 4, 0><<<waves / 8, 256, 0, stream>>>(
        act_bf, sfc3_bf, (void*)out, nullptr, 0, TT, DD, SS, SS, SS, DD);
  }

  // 5) MoE: dense per-expert compute weighted by dispatch (matches reference math)
  for (int e = 0; e < EE; ++e) {
    const bf16* w1e = w1_bf + (size_t)e * HH * DD;  // [H, D]
    const bf16* w2e = w2_bf + (size_t)e * DD * HH;  // [D, H]
    // h = silu(x @ w1e^T) * dispatch[:, e]   [T, H] bf16
    {
      int waves = (TT / 64) * (HH / 64);
      gemm_bf16<4, 4, 2><<<waves / 8, 256, 0, stream>>>(
          x_bf, w1e, (void*)h_bf, dispatch + e, EE, TT, HH, DD, DD, DD, HH);
    }
    // out += h @ w2e^T
    {
      int waves = (TT / 64) * (DD / 64);
      gemm_bf16<4, 4, 1><<<waves / 8, 256, 0, stream>>>(
          h_bf, w2e, (void*)out, nullptr, 0, TT, DD, HH, HH, HH, DD);
    }
  }
}
